// EinopsAttention_489626272085
// MI455X (gfx1250) — matrix operand order repaired and checked by reading, not compiled, not run on hardware
//
#include <hip/hip_runtime.h>

#define Bv   4
#define Nv   1024
#define Dv   1024
#define Hv   16
#define HDv  64
#define BHv  (Bv*Hv)
#define SCALEv 0.125f   // HD^-0.5

typedef __attribute__((ext_vector_type(16))) __bf16 v16bf;
typedef __attribute__((ext_vector_type(8)))  __bf16 v8bf;
typedef __attribute__((ext_vector_type(8)))  float  v8f;

static __device__ __forceinline__ v8f wmma_bf16(v16bf a, v16bf b, v8f c) {
  // D = A(16x32 bf16) x B(32x16 bf16) + C(16x16 f32)
  return __builtin_amdgcn_wmma_f32_16x16x32_bf16(false, a, false, b, (short)0, c, false, false);
}
static __device__ __forceinline__ v16bf cat8(v8bf lo, v8bf hi) {
  return __builtin_shufflevector(lo, hi, 0,1,2,3,4,5,6,7,8,9,10,11,12,13,14,15);
}
static __device__ __forceinline__ v16bf pack16(float4 a, float4 b, float4 c, float4 d) {
  v16bf r;
  r[0]=(__bf16)a.x;  r[1]=(__bf16)a.y;  r[2]=(__bf16)a.z;  r[3]=(__bf16)a.w;
  r[4]=(__bf16)b.x;  r[5]=(__bf16)b.y;  r[6]=(__bf16)b.z;  r[7]=(__bf16)b.w;
  r[8]=(__bf16)c.x;  r[9]=(__bf16)c.y;  r[10]=(__bf16)c.z; r[11]=(__bf16)c.w;
  r[12]=(__bf16)d.x; r[13]=(__bf16)d.y; r[14]=(__bf16)d.z; r[15]=(__bf16)d.w;
  return r;
}

// ---------------------------------------------------------------- fp32 -> bf16
__global__ void cvt_f32_bf16(const float* __restrict__ in, __bf16* __restrict__ out, int n8) {
  int i = blockIdx.x * blockDim.x + threadIdx.x;
  if (i >= n8) return;
  const float4* p = (const float4*)in + (size_t)i * 2;
  float4 a = p[0], b = p[1];
  v8bf o;
  o[0]=(__bf16)a.x; o[1]=(__bf16)a.y; o[2]=(__bf16)a.z; o[3]=(__bf16)a.w;
  o[4]=(__bf16)b.x; o[5]=(__bf16)b.y; o[6]=(__bf16)b.z; o[7]=(__bf16)b.w;
  ((v8bf*)out)[i] = o;
}

// --------------------------------------------------- QKV projection (bf16 WMMA)
// out = x @ W.T + b ; mode 0:q (scaled, [B,H,N,HD]) 1:k ([B,H,N,HD]) 2:v (transposed [B,H,HD,N])
__global__ void __launch_bounds__(256)
qkv_proj_kernel(const __bf16* __restrict__ xq, const __bf16* __restrict__ xk,
                const __bf16* __restrict__ xv,
                const __bf16* __restrict__ wq, const __bf16* __restrict__ wk,
                const __bf16* __restrict__ wv,
                const float* __restrict__ bq, const float* __restrict__ bk,
                const float* __restrict__ bv,
                __bf16* __restrict__ qh, __bf16* __restrict__ kh,
                __bf16* __restrict__ vhT) {
  const int mode = blockIdx.y;
  const __bf16* x    = (mode == 0) ? xq : (mode == 1) ? xk : xv;
  const __bf16* w    = (mode == 0) ? wq : (mode == 1) ? wk : wv;
  const float*  bias = (mode == 0) ? bq : (mode == 1) ? bk : bv;

  const int lane = threadIdx.x & 31, wid = threadIdx.x >> 5;
  const int gw = blockIdx.x * 8 + wid;          // 4096 wave tiles: 128 x 32
  const int tn = gw & 31, tm = gw >> 5;
  const int m0 = tm * 32, n0 = tn * 32;
  const int r = lane & 15, c = lane >> 4;

  const __bf16* a0p = x + (size_t)(m0 + r) * Dv;
  const __bf16* a1p = x + (size_t)(m0 + 16 + r) * Dv;
  const __bf16* b0p = w + (size_t)(n0 + r) * Dv;
  const __bf16* b1p = w + (size_t)(n0 + 16 + r) * Dv;

  v8f acc[2][2] = {};
  for (int kk = 0; kk < Dv; kk += 32) {
    v16bf A0 = cat8(*(const v8bf*)(a0p + kk + c*16), *(const v8bf*)(a0p + kk + c*16 + 8));
    v16bf A1 = cat8(*(const v8bf*)(a1p + kk + c*16), *(const v8bf*)(a1p + kk + c*16 + 8));
    v16bf B0 = *(const v16bf*)(b0p + kk + c*16);
    v16bf B1 = *(const v16bf*)(b1p + kk + c*16);
    acc[0][0] = wmma_bf16(A0, B0, acc[0][0]);
    acc[0][1] = wmma_bf16(A0, B1, acc[0][1]);
    acc[1][0] = wmma_bf16(A1, B0, acc[1][0]);
    acc[1][1] = wmma_bf16(A1, B1, acc[1][1]);
  }
  const float scale = (mode == 0) ? SCALEv : 1.0f;
  const float bias0 = bias[n0 + r], bias1 = bias[n0 + 16 + r];
  #pragma unroll
  for (int i = 0; i < 2; ++i)
    #pragma unroll
    for (int j = 0; j < 2; ++j) {
      float bj = j ? bias1 : bias0;
      #pragma unroll
      for (int p = 0; p < 8; ++p) {
        int tok = m0 + i*16 + p + 8*c;
        int f   = n0 + j*16 + r;
        int b_ = tok >> 10, q_ = tok & (Nv - 1);
        int h_ = f >> 6,    d_ = f & (HDv - 1);
        float val = (acc[i][j][p] + bj) * scale;
        if (mode == 2)
          vhT[((size_t)(b_*Hv + h_)*HDv + d_)*Nv + q_] = (__bf16)val;
        else if (mode == 1)
          kh[((size_t)(b_*Hv + h_)*Nv + q_)*HDv + d_] = (__bf16)val;
        else
          qh[((size_t)(b_*Hv + h_)*Nv + q_)*HDv + d_] = (__bf16)val;
      }
    }
}

// --------------------------- per-q rel-position scores: GEMM M=64(bh) N=1024(k) K=64(d)
__global__ void __launch_bounds__(256)
rel_score_kernel(const __bf16* __restrict__ qh, const float* __restrict__ rel,
                 __bf16* __restrict__ srel) {
  const int qidx = blockIdx.x;
  const int lane = threadIdx.x & 31, wid = threadIdx.x >> 5;
  const int mt = wid & 3;                               // bh tile of 16
  const int kbase = blockIdx.y * 256 + (wid >> 2) * 128;
  const int r = lane & 15, c = lane >> 4;

  // A fragments: 16 bh rows of qh at this q (K = d, two 32-steps)
  const __bf16* rq = qh + ((size_t)(mt*16 + r) * Nv + qidx) * HDv;
  v16bf A0 = cat8(*(const v8bf*)(rq + c*16),      *(const v8bf*)(rq + 8 + c*16));
  v16bf A1 = cat8(*(const v8bf*)(rq + 32 + c*16), *(const v8bf*)(rq + 40 + c*16));

  for (int kt = 0; kt < 8; ++kt) {
    const int k0 = kbase + kt * 16;
    const float* rr = rel + ((size_t)qidx * Nv + (k0 + r)) * HDv + c*16;
    __builtin_prefetch((const void*)(rr + 16 * HDv), 0, 0);   // next k-tile stream
    const float4* rp  = (const float4*)rr;
    const float4* rp2 = (const float4*)(rr + 32);
    v16bf B0 = pack16(rp[0],  rp[1],  rp[2],  rp[3]);   // d =  c*16 .. +16
    v16bf B1 = pack16(rp2[0], rp2[1], rp2[2], rp2[3]);  // d = 32+c*16 .. +16
    v8f acc = {};
    acc = wmma_bf16(A0, B0, acc);
    acc = wmma_bf16(A1, B1, acc);
    #pragma unroll
    for (int p = 0; p < 8; ++p) {
      int bh = mt*16 + p + 8*c;
      srel[((size_t)bh * Nv + qidx) * Nv + k0 + r] = (__bf16)acc[p];
    }
  }
}

// -------------------- attention: scores (WMMA) -> LDS -> exact softmax -> PV (WMMA)
#define ATTN_SMEM (4 * 16 * Nv * 4 + 4 * 16 * 4)
__global__ void __launch_bounds__(128)
attn_kernel(const __bf16* __restrict__ qh, const __bf16* __restrict__ kh,
            const __bf16* __restrict__ vhT, const __bf16* __restrict__ srel,
            __bf16* __restrict__ Obuf) {
  extern __shared__ char smem[];
  const int lane = threadIdx.x & 31, wid = threadIdx.x >> 5;
  float* Srow  = (float*)smem + (size_t)wid * 16 * Nv;            // 16 x 1024 f32 per wave
  float* stats = (float*)(smem + (size_t)4 * 16 * Nv * 4) + wid * 16;
  const int job = blockIdx.x * 4 + wid;                           // 4096 jobs = 64 bh x 64 qtiles
  const int bh = job >> 6;
  const int q0 = (job & 63) * 16;
  const int r = lane & 15, c = lane >> 4;

  const __bf16* rq = qh + ((size_t)bh * Nv + q0 + r) * HDv;
  v16bf QA0 = cat8(*(const v8bf*)(rq + c*16),      *(const v8bf*)(rq + 8 + c*16));
  v16bf QA1 = cat8(*(const v8bf*)(rq + 32 + c*16), *(const v8bf*)(rq + 40 + c*16));

  // ---- pass 1: S = Q K^T + S_rel, parked in LDS
  for (int kt = 0; kt < Nv / 16; ++kt) {
    const int k0 = kt * 16;
    const __bf16* rk = kh + ((size_t)bh * Nv + k0 + r) * HDv;
    __builtin_prefetch((const void*)(rk + 16 * HDv), 0, 0);
    v16bf KB0 = *(const v16bf*)(rk + c*16);
    v16bf KB1 = *(const v16bf*)(rk + 32 + c*16);
    v8f acc = {};
    acc = wmma_bf16(QA0, KB0, acc);
    acc = wmma_bf16(QA1, KB1, acc);
    const __bf16* sr = srel + ((size_t)bh * Nv + q0 + 8*c) * Nv + k0 + r;
    #pragma unroll
    for (int p = 0; p < 8; ++p) {
      float rv = (float)sr[(size_t)p * Nv];
      Srow[(p + 8*c) * Nv + k0 + r] = acc[p] + rv;   // C layout: row p+8c, col lane&15
    }
  }

  // ---- pass 2: exact softmax per row (lane r owns row r, halves split columns)
  {
    float* sp = Srow + r * Nv + c * 512;
    float m = -3.0e38f;
    #pragma unroll 8
    for (int i = 0; i < 512; ++i) m = fmaxf(m, sp[i]);
    m = fmaxf(m, __shfl_xor(m, 16, 32));
    float l = 0.0f;
    #pragma unroll 8
    for (int i = 0; i < 512; ++i) { float e = __expf(sp[i] - m); sp[i] = e; l += e; }
    l += __shfl_xor(l, 16, 32);
    if (c == 0) stats[r] = 1.0f / l;
  }

  // ---- pass 3: O = P x V (A-fragments rebuilt from LDS, B from transposed V)
  v8f o0 = {}, o1 = {}, o2 = {}, o3 = {};
  for (int kk = 0; kk < Nv; kk += 32) {
    const float* ap = Srow + r * Nv + kk + c*16;
    float4 f0 = *(const float4*)(ap);
    float4 f1 = *(const float4*)(ap + 4);
    float4 f2 = *(const float4*)(ap + 8);
    float4 f3 = *(const float4*)(ap + 12);
    v16bf PA = pack16(f0, f1, f2, f3);
    const __bf16* rv0 = vhT + ((size_t)bh * HDv + r) * Nv + kk + c*16;
    __builtin_prefetch((const void*)(rv0 + 32), 0, 0);
    o0 = wmma_bf16(PA, *(const v16bf*)(rv0),                  o0);
    o1 = wmma_bf16(PA, *(const v16bf*)(rv0 + (size_t)16*Nv),  o1);
    o2 = wmma_bf16(PA, *(const v16bf*)(rv0 + (size_t)32*Nv),  o2);
    o3 = wmma_bf16(PA, *(const v16bf*)(rv0 + (size_t)48*Nv),  o3);
  }
  const int b_ = bh >> 4, h_ = bh & (Hv - 1);
  #pragma unroll
  for (int p = 0; p < 8; ++p) {
    float linv = stats[p + 8*c];
    size_t row = (size_t)(b_ * Nv + q0 + p + 8*c) * Dv + h_ * HDv + r;
    Obuf[row]      = (__bf16)(o0[p] * linv);
    Obuf[row + 16] = (__bf16)(o1[p] * linv);
    Obuf[row + 32] = (__bf16)(o2[p] * linv);
    Obuf[row + 48] = (__bf16)(o3[p] * linv);
  }
}

// --------------------------------------------- output projection (fp32 result)
__global__ void __launch_bounds__(256)
out_proj_kernel(const __bf16* __restrict__ O, const __bf16* __restrict__ wo,
                const float* __restrict__ bo, float* __restrict__ out) {
  const int lane = threadIdx.x & 31, wid = threadIdx.x >> 5;
  const int gw = blockIdx.x * 8 + wid;
  const int tn = gw & 31, tm = gw >> 5;
  const int m0 = tm * 32, n0 = tn * 32;
  const int r = lane & 15, c = lane >> 4;
  const __bf16* a0p = O  + (size_t)(m0 + r) * Dv;
  const __bf16* a1p = O  + (size_t)(m0 + 16 + r) * Dv;
  const __bf16* b0p = wo + (size_t)(n0 + r) * Dv;
  const __bf16* b1p = wo + (size_t)(n0 + 16 + r) * Dv;
  v8f acc[2][2] = {};
  for (int kk = 0; kk < Dv; kk += 32) {
    v16bf A0 = cat8(*(const v8bf*)(a0p + kk + c*16), *(const v8bf*)(a0p + kk + c*16 + 8));
    v16bf A1 = cat8(*(const v8bf*)(a1p + kk + c*16), *(const v8bf*)(a1p + kk + c*16 + 8));
    v16bf B0 = *(const v16bf*)(b0p + kk + c*16);
    v16bf B1 = *(const v16bf*)(b1p + kk + c*16);
    acc[0][0] = wmma_bf16(A0, B0, acc[0][0]);
    acc[0][1] = wmma_bf16(A0, B1, acc[0][1]);
    acc[1][0] = wmma_bf16(A1, B0, acc[1][0]);
    acc[1][1] = wmma_bf16(A1, B1, acc[1][1]);
  }
  const float bias0 = bo[n0 + r], bias1 = bo[n0 + 16 + r];
  #pragma unroll
  for (int i = 0; i < 2; ++i)
    #pragma unroll
    for (int j = 0; j < 2; ++j) {
      float bj = j ? bias1 : bias0;
      #pragma unroll
      for (int p = 0; p < 8; ++p)
        out[(size_t)(m0 + i*16 + p + 8*c) * Dv + n0 + j*16 + r] = acc[i][j][p] + bj;
    }
}

// ------------------------------------------------------------------ launcher
extern "C" void kernel_launch(void* const* d_in, const int* in_sizes, int n_in,
                              void* d_out, int out_size, void* d_ws, size_t ws_size,
                              hipStream_t stream) {
  (void)in_sizes; (void)n_in; (void)out_size; (void)ws_size;
  const float* q   = (const float*)d_in[0];
  const float* k   = (const float*)d_in[1];
  const float* v   = (const float*)d_in[2];
  const float* rel = (const float*)d_in[3];
  const float* Wq  = (const float*)d_in[4];
  const float* bq  = (const float*)d_in[5];
  const float* Wk  = (const float*)d_in[6];
  const float* bk  = (const float*)d_in[7];
  const float* Wv  = (const float*)d_in[8];
  const float* bv  = (const float*)d_in[9];
  const float* Wo  = (const float*)d_in[10];
  const float* bo  = (const float*)d_in[11];

  char* ws = (char*)d_ws;
  constexpr size_t SZ_X = (size_t)Bv * Nv * Dv * 2;       // 8 MiB
  constexpr size_t SZ_W = (size_t)Dv * Dv * 2;            // 2 MiB
  constexpr size_t SZ_H = (size_t)BHv * Nv * HDv * 2;     // 8 MiB
  constexpr size_t SZ_S = (size_t)BHv * Nv * Nv * 2;      // 128 MiB
  __bf16* xq   = (__bf16*)(ws);
  __bf16* xk   = (__bf16*)(ws + SZ_X);
  __bf16* xv   = (__bf16*)(ws + 2*SZ_X);
  __bf16* wq   = (__bf16*)(ws + 3*SZ_X);
  __bf16* wk   = (__bf16*)(ws + 3*SZ_X + SZ_W);
  __bf16* wv   = (__bf16*)(ws + 3*SZ_X + 2*SZ_W);
  __bf16* wo   = (__bf16*)(ws + 3*SZ_X + 3*SZ_W);
  __bf16* qh   = (__bf16*)(ws + 3*SZ_X + 4*SZ_W);
  __bf16* kh   = (__bf16*)(ws + 3*SZ_X + 4*SZ_W + SZ_H);
  __bf16* vhT  = (__bf16*)(ws + 3*SZ_X + 4*SZ_W + 2*SZ_H);
  __bf16* srel = (__bf16*)(ws + 3*SZ_X + 4*SZ_W + 3*SZ_H);
  __bf16* Obuf = (__bf16*)(ws + 3*SZ_X + 4*SZ_W + 3*SZ_H + SZ_S);

  // fp32 -> bf16 staging
  cvt_f32_bf16<<<2048, 256, 0, stream>>>(q,  xq, (Bv*Nv*Dv)/8);
  cvt_f32_bf16<<<2048, 256, 0, stream>>>(k,  xk, (Bv*Nv*Dv)/8);
  cvt_f32_bf16<<<2048, 256, 0, stream>>>(v,  xv, (Bv*Nv*Dv)/8);
  cvt_f32_bf16<<<512,  256, 0, stream>>>(Wq, wq, (Dv*Dv)/8);
  cvt_f32_bf16<<<512,  256, 0, stream>>>(Wk, wk, (Dv*Dv)/8);
  cvt_f32_bf16<<<512,  256, 0, stream>>>(Wv, wv, (Dv*Dv)/8);
  cvt_f32_bf16<<<512,  256, 0, stream>>>(Wo, wo, (Dv*Dv)/8);

  qkv_proj_kernel<<<dim3(512, 3), 256, 0, stream>>>(xq, xk, xv, wq, wk, wv,
                                                    bq, bk, bv, qh, kh, vhT);
  rel_score_kernel<<<dim3(Nv, 4), 256, 0, stream>>>(qh, rel, srel);
  attn_kernel<<<1024, 128, ATTN_SMEM, stream>>>(qh, kh, vhT, srel, Obuf);
  out_proj_kernel<<<512, 256, 0, stream>>>(Obuf, wo, bo, (float*)d_out);
}
